// GraphSAGE2_46488726012230
// MI455X (gfx1250) — matrix operand order repaired and checked by reading, not compiled, not run on hardware
//
#include <hip/hip_runtime.h>
#include <hip/hip_bf16.h>

// ---------------------------------------------------------------------------
// GraphSAGE autoencoder on MI455X (gfx1250, wave32, WMMA).
//  - bf16 operands everywhere in GEMMs; v_wmma_f32_16x16x32_bf16, f32 acc.
//  - Each wave computes a 32x64 tile: 8 WMMAs per K-step from 2 A-frags and
//    4 B-frags loaded into distinct registers (enables partial loadcnt waits
//    and load/WMMA overlap instead of full s_wait_loadcnt 0x0 serialization).
//  - Edge aggregation: bf16 gather + f32 atomic scatter (L2-resident).
// ---------------------------------------------------------------------------

typedef __attribute__((ext_vector_type(16))) __bf16 v16bf;
typedef __attribute__((ext_vector_type(8)))  __bf16 v8bf;
typedef __attribute__((ext_vector_type(8)))  float  v8f;

#define N_NODES 20000
#define N_EDGES 320000
#define D_IN    128
#define D_HID   256

__device__ __forceinline__ __bf16 cvt_bf16(float f) {
    unsigned u = __builtin_bit_cast(unsigned, f);
    u += 0x7FFFu + ((u >> 16) & 1u);               // round-to-nearest-even
    unsigned short h = (unsigned short)(u >> 16);
    return __builtin_bit_cast(__bf16, h);
}

__device__ __forceinline__ float bf2f(__bf16 h) {
    unsigned short b = __builtin_bit_cast(unsigned short, h);
    unsigned u = ((unsigned)b) << 16;
    return __builtin_bit_cast(float, u);
}

__device__ __forceinline__ v16bf load_frag(const __bf16* __restrict__ p0,
                                           const __bf16* __restrict__ p1) {
    v8bf lo = *(const v8bf*)p0;
    v8bf hi = *(const v8bf*)p1;
    return __builtin_shufflevector(lo, hi,
               0,1,2,3,4,5,6,7,8,9,10,11,12,13,14,15);
}

// acc[j*2+r] += A-tile(rows arow_r, 16xK) x W-tile(Kx16 col-block j).
// A: [M,K] bf16 row-major; W: [O,K] bf16 row-major (B[k][o] = W[o][k]).
__device__ __forceinline__ void gemm_pass(v8f acc[8],
                                          const __bf16* __restrict__ A,
                                          const __bf16* __restrict__ W,
                                          int K, int arow0, int arow1,
                                          int obase, int l16, int half) {
    const __bf16* Ar0 = A + (long long)arow0 * K;
    const __bf16* Ar1 = A + (long long)arow1 * K;
    const __bf16* Wp0 = W + (long long)(obase +  0 + l16) * K;
    const __bf16* Wp1 = W + (long long)(obase + 16 + l16) * K;
    const __bf16* Wp2 = W + (long long)(obase + 32 + l16) * K;
    const __bf16* Wp3 = W + (long long)(obase + 48 + l16) * K;
    for (int k0 = 0; k0 < K; k0 += 32) {
        // A fragment: lane -> row, half selects K 0..15/16..31, contiguous.
        int ka = k0 + half * 16;
        // B fragment: lane -> col l16, half selects K 0..15/16..31, contiguous.
        int kb = k0 + half * 16;
        v16bf a0 = load_frag(Ar0 + ka, Ar0 + ka + 8);
        v16bf a1 = load_frag(Ar1 + ka, Ar1 + ka + 8);
        v16bf b0 = load_frag(Wp0 + kb, Wp0 + kb + 8);
        v16bf b1 = load_frag(Wp1 + kb, Wp1 + kb + 8);
        v16bf b2 = load_frag(Wp2 + kb, Wp2 + kb + 8);
        v16bf b3 = load_frag(Wp3 + kb, Wp3 + kb + 8);
        acc[0] = __builtin_amdgcn_wmma_f32_16x16x32_bf16(false, a0, false, b0, (short)0, acc[0], false, false);
        acc[1] = __builtin_amdgcn_wmma_f32_16x16x32_bf16(false, a1, false, b0, (short)0, acc[1], false, false);
        acc[2] = __builtin_amdgcn_wmma_f32_16x16x32_bf16(false, a0, false, b1, (short)0, acc[2], false, false);
        acc[3] = __builtin_amdgcn_wmma_f32_16x16x32_bf16(false, a1, false, b1, (short)0, acc[3], false, false);
        acc[4] = __builtin_amdgcn_wmma_f32_16x16x32_bf16(false, a0, false, b2, (short)0, acc[4], false, false);
        acc[5] = __builtin_amdgcn_wmma_f32_16x16x32_bf16(false, a1, false, b2, (short)0, acc[5], false, false);
        acc[6] = __builtin_amdgcn_wmma_f32_16x16x32_bf16(false, a0, false, b3, (short)0, acc[6], false, false);
        acc[7] = __builtin_amdgcn_wmma_f32_16x16x32_bf16(false, a1, false, b3, (short)0, acc[7], false, false);
    }
}

// out[m][o] = sum_k A1[m][k]*W1[o][k] + (A2? sum_k A2[m][k]*W2[o][k] :0) + bias[o]
// grid = (M/32, O/64), block = 32 (one wave -> 32x64 tile).
__global__ __launch_bounds__(32)
void gemm_wmma(const __bf16* __restrict__ A1, const __bf16* __restrict__ W1,
               const __bf16* __restrict__ A2, const __bf16* __restrict__ W2,
               const float* __restrict__ bias, void* __restrict__ out,
               int K, int O, int out_is_bf16) {
    int mbase = blockIdx.x * 32;
    int obase = blockIdx.y * 64;
    int lane = threadIdx.x;
    int half = lane >> 4, l16 = lane & 15;
    int arow0 = mbase + l16;
    int arow1 = mbase + 16 + l16;

    v8f acc[8] = {};
    gemm_pass(acc, A1, W1, K, arow0, arow1, obase, l16, half);
    if (A2) gemm_pass(acc, A2, W2, K, arow0, arow1, obase, l16, half);

    // C/D layout: VGPR v -> row (v + 8*half), column l16.
    #pragma unroll
    for (int j = 0; j < 4; ++j) {
        int ocol = obase + j * 16 + l16;
        float bv = bias ? bias[ocol] : 0.0f;
        #pragma unroll
        for (int r = 0; r < 2; ++r) {
            #pragma unroll
            for (int v = 0; v < 8; ++v) {
                int m = mbase + r * 16 + v + 8 * half;
                float res = acc[j * 2 + r][v] + bv;
                if (out_is_bf16)
                    ((__bf16*)out)[(long long)m * O + ocol] = cvt_bf16(res);
                else
                    ((float*)out)[(long long)m * O + ocol] = res;
            }
        }
    }
}

__global__ void f32_to_bf16_kernel(const float* __restrict__ in,
                                   __bf16* __restrict__ out, long long n) {
    long long i = (long long)blockIdx.x * blockDim.x + threadIdx.x;
    if (i < n) out[i] = cvt_bf16(in[i]);
}

__global__ void deg_kernel(const long long* __restrict__ dst,
                           float* __restrict__ deg, int E) {
    int e = blockIdx.x * blockDim.x + threadIdx.x;
    if (e < E) atomicAdd(&deg[(int)dst[e]], 1.0f);
}

__global__ void rdeg_kernel(const float* __restrict__ deg,
                            float* __restrict__ rdeg, int n) {
    int i = blockIdx.x * blockDim.x + threadIdx.x;
    if (i < n) rdeg[i] = 1.0f / fmaxf(deg[i], 1.0f);
}

// One block per edge: agg[dst,:] += (float)z16[src,:]
__global__ void scatter_add(const __bf16* __restrict__ z16,
                            const long long* __restrict__ src,
                            const long long* __restrict__ dst,
                            float* __restrict__ agg, int d) {
    int e = blockIdx.x;
    long long s = src[e], t = dst[e];
    const __bf16* zs = z16 + s * d;
    float* at = agg + t * d;
    for (int f = threadIdx.x; f < d; f += blockDim.x)
        atomicAdd(&at[f], bf2f(zs[f]));
}

// mean16[i] = bf16(agg[i] * rdeg[i >> dlog2])   (d = 1<<dlog2)
__global__ void mean_prep(const float* __restrict__ agg,
                          const float* __restrict__ rdeg,
                          __bf16* __restrict__ out, long long total, int dlog2) {
    long long i = (long long)blockIdx.x * blockDim.x + threadIdx.x;
    if (i < total) {
        int r = (int)(i >> dlog2);
        out[i] = cvt_bf16(agg[i] * rdeg[r]);
    }
}

// Coalesced per-column partial sums: thread = column (d==256), block strides rows.
__global__ __launch_bounds__(256)
void bn_partial(const float* __restrict__ z, float* __restrict__ s,
                float* __restrict__ s2, int n, int d) {
    int c = threadIdx.x;
    float a = 0.f, b = 0.f;
    for (int r = blockIdx.x; r < n; r += gridDim.x) {
        float v = z[(long long)r * d + c];
        a += v; b += v * v;
    }
    atomicAdd(&s[c], a);
    atomicAdd(&s2[c], b);
}

__global__ void bn_finalize(const float* __restrict__ s,
                            const float* __restrict__ s2,
                            float* __restrict__ mean, float* __restrict__ rstd,
                            int n, int d) {
    int c = blockIdx.x * blockDim.x + threadIdx.x;
    if (c < d) {
        float m = s[c] / n;
        float var = s2[c] / n - m * m;
        mean[c] = m;
        rstd[c] = rsqrtf(var + 1e-5f);
    }
}

// z16out = bf16(relu((z - mean) * rstd * g + b)), c = i & (d-1)
__global__ void bn_relu_bf16(const float* __restrict__ z,
                             const float* __restrict__ mean,
                             const float* __restrict__ rstd,
                             const float* __restrict__ g,
                             const float* __restrict__ b,
                             __bf16* __restrict__ out, long long total, int dmask) {
    long long i = (long long)blockIdx.x * blockDim.x + threadIdx.x;
    if (i < total) {
        int c = (int)(i & dmask);
        float v = (z[i] - mean[c]) * rstd[c] * g[c] + b[c];
        out[i] = cvt_bf16(v > 0.f ? v : 0.f);
    }
}

static inline unsigned cdiv(long long a, long long b) {
    return (unsigned)((a + b - 1) / b);
}

extern "C" void kernel_launch(void* const* d_in, const int* in_sizes, int n_in,
                              void* d_out, int out_size, void* d_ws, size_t ws_size,
                              hipStream_t stream) {
    (void)in_sizes; (void)n_in; (void)out_size; (void)ws_size;
    const float*     x   = (const float*)d_in[0];
    const long long* ei  = (const long long*)d_in[1];   // int64 [2, E]
    const float* enc_W = (const float*)d_in[2];
    const float* enc_b = (const float*)d_in[3];
    const float* Wl0 = (const float*)d_in[4];  const float* bl0 = (const float*)d_in[5];
    const float* Wr0 = (const float*)d_in[6];
    const float* Wl1 = (const float*)d_in[7];  const float* bl1 = (const float*)d_in[8];
    const float* Wr1 = (const float*)d_in[9];
    const float* Wl2 = (const float*)d_in[10]; const float* bl2 = (const float*)d_in[11];
    const float* Wr2 = (const float*)d_in[12];
    const float* Wl3 = (const float*)d_in[13]; const float* bl3 = (const float*)d_in[14];
    const float* Wr3 = (const float*)d_in[15];
    const float* g0 = (const float*)d_in[16]; const float* b0 = (const float*)d_in[17];
    const float* g1 = (const float*)d_in[18]; const float* b1 = (const float*)d_in[19];
    const float* g2 = (const float*)d_in[20]; const float* b2 = (const float*)d_in[21];
    const float* dec_W = (const float*)d_in[22];
    const float* dec_b = (const float*)d_in[23];
    float* out = (float*)d_out;

    const int N = N_NODES, E = N_EDGES, DI = D_IN, DH = D_HID;
    const long long nh = (long long)N * DH;     // 5.12M
    const long long ni = (long long)N * DI;     // 2.56M

    // ---- workspace layout ----
    char* p = (char*)d_ws;
    __bf16* z16A  = (__bf16*)p;            p += nh * 2;
    __bf16* z16B  = (__bf16*)p;            p += nh * 2;
    __bf16* agg16 = (__bf16*)p;            p += nh * 2;
    __bf16* x16   = (__bf16*)p;            p += ni * 2;
    float*  tmp   = (float*)p;             p += nh * 4;
    float*  agg   = (float*)p;             p += nh * 4;
    float*  deg   = (float*)p;             p += N * 4;
    float*  rdeg  = (float*)p;             p += N * 4;
    float*  bnS   = (float*)p;             p += DH * 4;
    float*  bnS2  = (float*)p;             p += DH * 4;
    float*  bnM   = (float*)p;             p += DH * 4;
    float*  bnR   = (float*)p;             p += DH * 4;
    __bf16* wpool = (__bf16*)p;
    __bf16* enc16 = wpool;                 wpool += DI * DI;
    __bf16* Wl0_16 = wpool;                wpool += DH * DI;
    __bf16* Wr0_16 = wpool;                wpool += DH * DI;
    __bf16* Wl1_16 = wpool;                wpool += DH * DH;
    __bf16* Wr1_16 = wpool;                wpool += DH * DH;
    __bf16* Wl2_16 = wpool;                wpool += DH * DH;
    __bf16* Wr2_16 = wpool;                wpool += DH * DH;
    __bf16* Wl3_16 = wpool;                wpool += DI * DH;
    __bf16* Wr3_16 = wpool;                wpool += DI * DH;
    __bf16* dec16  = wpool;                wpool += DI * DI;

    const long long* src = ei;
    const long long* dst = ei + E;

    // ---- weight + input bf16 conversion ----
    f32_to_bf16_kernel<<<cdiv(DI*DI,256), 256, 0, stream>>>(enc_W, enc16, DI*DI);
    f32_to_bf16_kernel<<<cdiv(DH*DI,256), 256, 0, stream>>>(Wl0, Wl0_16, DH*DI);
    f32_to_bf16_kernel<<<cdiv(DH*DI,256), 256, 0, stream>>>(Wr0, Wr0_16, DH*DI);
    f32_to_bf16_kernel<<<cdiv(DH*DH,256), 256, 0, stream>>>(Wl1, Wl1_16, DH*DH);
    f32_to_bf16_kernel<<<cdiv(DH*DH,256), 256, 0, stream>>>(Wr1, Wr1_16, DH*DH);
    f32_to_bf16_kernel<<<cdiv(DH*DH,256), 256, 0, stream>>>(Wl2, Wl2_16, DH*DH);
    f32_to_bf16_kernel<<<cdiv(DH*DH,256), 256, 0, stream>>>(Wr2, Wr2_16, DH*DH);
    f32_to_bf16_kernel<<<cdiv(DI*DH,256), 256, 0, stream>>>(Wl3, Wl3_16, (long long)DI*DH);
    f32_to_bf16_kernel<<<cdiv(DI*DH,256), 256, 0, stream>>>(Wr3, Wr3_16, (long long)DI*DH);
    f32_to_bf16_kernel<<<cdiv(DI*DI,256), 256, 0, stream>>>(dec_W, dec16, DI*DI);
    f32_to_bf16_kernel<<<cdiv(ni,256), 256, 0, stream>>>(x, x16, ni);

    // ---- degrees (same for all layers) ----
    hipMemsetAsync(deg, 0, N * sizeof(float), stream);
    deg_kernel<<<cdiv(E,256), 256, 0, stream>>>(dst, deg, E);
    rdeg_kernel<<<cdiv(N,256), 256, 0, stream>>>(deg, rdeg, N);

    // ---- encoder: z16A = bf16(x @ enc_W^T + enc_b)  [N,128] ----
    gemm_wmma<<<dim3(N/32, DI/64), 32, 0, stream>>>(
        x16, enc16, nullptr, nullptr, enc_b, z16A, DI, DI, 1);

    // ---- SAGE layer 0: 128 -> 256 (z16A -> z16B) ----
    hipMemsetAsync(agg, 0, (size_t)ni * 4, stream);
    scatter_add<<<E, 128, 0, stream>>>(z16A, src, dst, agg, DI);
    mean_prep<<<cdiv(ni,256), 256, 0, stream>>>(agg, rdeg, agg16, ni, 7);
    gemm_wmma<<<dim3(N/32, DH/64), 32, 0, stream>>>(
        agg16, Wl0_16, z16A, Wr0_16, bl0, tmp, DI, DH, 0);
    hipMemsetAsync(bnS, 0, DH * 4, stream);
    hipMemsetAsync(bnS2, 0, DH * 4, stream);
    bn_partial<<<256, 256, 0, stream>>>(tmp, bnS, bnS2, N, DH);
    bn_finalize<<<1, DH, 0, stream>>>(bnS, bnS2, bnM, bnR, N, DH);
    bn_relu_bf16<<<cdiv(nh,256), 256, 0, stream>>>(tmp, bnM, bnR, g0, b0, z16B, nh, DH - 1);

    // ---- SAGE layer 1: 256 -> 256 (z16B -> z16A) ----
    hipMemsetAsync(agg, 0, (size_t)nh * 4, stream);
    scatter_add<<<E, 256, 0, stream>>>(z16B, src, dst, agg, DH);
    mean_prep<<<cdiv(nh,256), 256, 0, stream>>>(agg, rdeg, agg16, nh, 8);
    gemm_wmma<<<dim3(N/32, DH/64), 32, 0, stream>>>(
        agg16, Wl1_16, z16B, Wr1_16, bl1, tmp, DH, DH, 0);
    hipMemsetAsync(bnS, 0, DH * 4, stream);
    hipMemsetAsync(bnS2, 0, DH * 4, stream);
    bn_partial<<<256, 256, 0, stream>>>(tmp, bnS, bnS2, N, DH);
    bn_finalize<<<1, DH, 0, stream>>>(bnS, bnS2, bnM, bnR, N, DH);
    bn_relu_bf16<<<cdiv(nh,256), 256, 0, stream>>>(tmp, bnM, bnR, g1, b1, z16A, nh, DH - 1);

    // ---- SAGE layer 2: 256 -> 256 (z16A -> z16B) ----
    hipMemsetAsync(agg, 0, (size_t)nh * 4, stream);
    scatter_add<<<E, 256, 0, stream>>>(z16A, src, dst, agg, DH);
    mean_prep<<<cdiv(nh,256), 256, 0, stream>>>(agg, rdeg, agg16, nh, 8);
    gemm_wmma<<<dim3(N/32, DH/64), 32, 0, stream>>>(
        agg16, Wl2_16, z16A, Wr2_16, bl2, tmp, DH, DH, 0);
    hipMemsetAsync(bnS, 0, DH * 4, stream);
    hipMemsetAsync(bnS2, 0, DH * 4, stream);
    bn_partial<<<256, 256, 0, stream>>>(tmp, bnS, bnS2, N, DH);
    bn_finalize<<<1, DH, 0, stream>>>(bnS, bnS2, bnM, bnR, N, DH);
    bn_relu_bf16<<<cdiv(nh,256), 256, 0, stream>>>(tmp, bnM, bnR, g2, b2, z16B, nh, DH - 1);

    // ---- SAGE layer 3: 256 -> 128, no BN (z16B -> z16A bf16) ----
    hipMemsetAsync(agg, 0, (size_t)nh * 4, stream);
    scatter_add<<<E, 256, 0, stream>>>(z16B, src, dst, agg, DH);
    mean_prep<<<cdiv(nh,256), 256, 0, stream>>>(agg, rdeg, agg16, nh, 8);
    gemm_wmma<<<dim3(N/32, DI/64), 32, 0, stream>>>(
        agg16, Wl3_16, z16B, Wr3_16, bl3, z16A, DH, DI, 1);

    // ---- decoder: out = z16A @ dec_W^T + dec_b  [N,128] f32 ----
    gemm_wmma<<<dim3(N/32, DI/64), 32, 0, stream>>>(
        z16A, dec16, nullptr, nullptr, dec_b, out, DI, DI, 0);
}
